// CMCM_30073361007267
// MI455X (gfx1250) — matrix operand order repaired and checked by reading, not compiled, not run on hardware
//
#include <hip/hip_runtime.h>
#include <cstdint>

// ---------------------------------------------------------------------------
// Model dims (fixed by reference)
// ---------------------------------------------------------------------------
#define BB   32
#define SS   512
#define DE_  256
#define DP_  128
#define EMB  256
#define WIN_ 9
#define NCLS 4
#define BS_  (BB * SS)          // 16384 positions
#define BIGF 1e10f

typedef __attribute__((ext_vector_type(16))) _Float16 v16h;
typedef __attribute__((ext_vector_type(8)))  float    v8f;

// ---------------------------------------------------------------------------
// Wave helpers (wave32)
// ---------------------------------------------------------------------------
__device__ __forceinline__ float wave_sum(float v) {
    #pragma unroll
    for (int o = 16; o; o >>= 1) v += __shfl_xor(v, o, 32);
    return v;
}
__device__ __forceinline__ float wave_max(float v) {
    #pragma unroll
    for (int o = 16; o; o >>= 1) v = fmaxf(v, __shfl_xor(v, o, 32));
    return v;
}

// ---------------------------------------------------------------------------
// Weight repack: f32 weights -> f16 WMMA B-fragments (packed half2 per dword).
// B fragment (32x16 f16): lane l holds column n = l&15 ; K half = (l>>4)*16 ;
// dword i holds K = khalf + 2i, 2i+1.
// headed>0: W is (H, K, hd) per-head stack; else plain row-major (K, N).
// ---------------------------------------------------------------------------
__global__ void repack_weight(const float* __restrict__ W, uint32_t* __restrict__ Wp,
                              int K, int N, int headed_hd) {
    int idx = blockIdx.x * blockDim.x + threadIdx.x;
    int total = (N / 16) * (K / 32) * 256;
    if (idx >= total) return;
    int i    = idx & 7;
    int lane = (idx >> 3) & 31;
    int ks   = (idx >> 8) % (K / 32);
    int tn   = (idx >> 8) / (K / 32);
    int n  = tn * 16 + (lane & 15);
    int k0 = ks * 32 + ((lane >> 4) << 4) + 2 * i;
    float w0, w1;
    if (headed_hd > 0) {
        int hd = headed_hd;
        int h = n / hd, e = n % hd;
        w0 = W[((size_t)h * K + k0) * hd + e];
        w1 = W[((size_t)h * K + k0 + 1) * hd + e];
    } else {
        w0 = W[(size_t)k0 * N + n];
        w1 = W[(size_t)(k0 + 1) * N + n];
    }
    union { _Float16 h[2]; uint32_t u; } cv;
    cv.h[0] = (_Float16)w0; cv.h[1] = (_Float16)w1;
    Wp[idx] = cv.u;
}

// ---------------------------------------------------------------------------
// Generic WMMA GEMM: Y[M,N] = X[M,K] @ W + bias, f32 in/out, f16 math.
// Register-blocked: each wave owns a 16-row strip and NT consecutive 16-col
// tiles; the A fragment is built once per K-step and reused across NT WMMAs
// (4x less A traffic / f16 conversion, dense back-to-back v_wmma issue).
// Block = 4 waves. grid.y * 4 waves * NT tiles * 16 cols == N.
// A fragment (16x32 f16): lane row = l&15, khalf=(l>>4)*16 ;
//   pair i: K = khalf + 2*i, 2*i+1.
// D: lane n = tile_n*16 + (l&15); rows m = tile_m*16 + (l>>4)*8 + r.
// ---------------------------------------------------------------------------
template <int K, int NT>
__global__ void wmma_gemm(const float* __restrict__ X, const uint32_t* __restrict__ Wp,
                          const float* __restrict__ bias, float* __restrict__ Y, int N) {
    constexpr int KSTEPS = K / 32;
    int wave = threadIdx.x >> 5;
    int lane = threadIdx.x & 31;
    int tm  = blockIdx.x;
    int tn0 = (blockIdx.y * 4 + wave) * NT;
    if (tn0 * 16 >= N) return;
    int hl  = lane >> 4;
    int row = tm * 16 + (lane & 15);
    const float* xrow = X + (size_t)row * K;

    v8f acc[NT];
    #pragma unroll
    for (int t = 0; t < NT; ++t) acc[t] = (v8f){};

    #pragma unroll
    for (int ks = 0; ks < KSTEPS; ++ks) {
        if (ks + 1 < KSTEPS)
            __builtin_prefetch(xrow + (ks + 1) * 32, 0, 3);   // global_prefetch_b8
        // --- A fragment: built once, reused by NT WMMAs ---
        v16h a;
        #pragma unroll
        for (int i = 0; i < 8; ++i) {
            int koff = (hl << 4) + (i << 1);
            float2 xv = *(const float2*)(xrow + ks * 32 + koff);
            a[2 * i]     = (_Float16)xv.x;
            a[2 * i + 1] = (_Float16)xv.y;
        }
        #pragma unroll
        for (int t = 0; t < NT; ++t) {
            const uint32_t* wl = Wp + ((size_t)(tn0 + t) * KSTEPS + ks) * 256 + lane * 8;
            v16h b;
            #pragma unroll
            for (int i = 0; i < 8; ++i) {
                union { uint32_t u; _Float16 h[2]; } cv; cv.u = wl[i];
                b[2 * i]     = cv.h[0];
                b[2 * i + 1] = cv.h[1];
            }
            acc[t] = __builtin_amdgcn_wmma_f32_16x16x32_f16(false, a, false, b,
                                                            (short)0, acc[t], false, false);
        }
    }
    #pragma unroll
    for (int t = 0; t < NT; ++t) {
        int n = (tn0 + t) * 16 + (lane & 15);
        float bv = bias ? bias[n] : 0.0f;
        #pragma unroll
        for (int r = 0; r < 8; ++r) {
            int m = tm * 16 + (hl << 3) + r;
            Y[(size_t)m * N + n] = acc[t][r] + bv;
        }
    }
}

// ---------------------------------------------------------------------------
// Attention: one wave per (b,s) position.
// intra=1: kv = [9-window of Kx/Vx, Kb/Vb token, Ka/Va token], q = Qa[s] (S rows)
// intra=0: kv = 9-window of Kx/Vx, q = Q[pos] (B*S rows)
// Out-of-range window rows -> bias rows bk / bv (zero row projected).
// ---------------------------------------------------------------------------
__global__ void attn_kernel(const float* __restrict__ Q,
                            const float* __restrict__ Kx, const float* __restrict__ Vx,
                            const float* __restrict__ Ka, const float* __restrict__ Va,
                            const float* __restrict__ Kb, const float* __restrict__ Vb,
                            const float* __restrict__ bk, const float* __restrict__ bv,
                            float* __restrict__ O, int Din, int hd, int intra, int S) {
    int wave = threadIdx.x >> 5;
    int lane = threadIdx.x & 31;
    int pos = blockIdx.x * (blockDim.x >> 5) + wave;
    if (pos >= BS_) return;
    int b = pos / S, s = pos % S;
    int H = Din / hd;
    int nkv = intra ? (WIN_ + 2) : WIN_;
    float scale = rsqrtf((float)hd);
    const float* qrow = Q + (size_t)(intra ? s : pos) * Din;

    for (int h = 0; h < H; ++h) {
        // --- scores, lane t < nkv ---
        float score = -1e30f;
        if (lane < nkv) {
            const float* kptr;
            if (lane < WIN_) {
                int p = s + lane - (WIN_ / 2);
                kptr = (p >= 0 && p < S) ? Kx + ((size_t)b * S + p) * Din : bk;
            } else if (lane == WIN_) {
                kptr = Kb + (size_t)s * Din;
            } else {
                kptr = Ka + (size_t)s * Din;
            }
            float acc = 0.0f;
            for (int e = 0; e < hd; ++e)
                acc += qrow[h * hd + e] * kptr[h * hd + e];
            score = acc * scale;
        }
        float m   = wave_max(score);
        float ex  = (lane < nkv) ? __expf(score - m) : 0.0f;
        float sum = wave_sum(ex);
        float w   = ex / sum;
        // --- weighted value sum, lane e < hd ---
        float o = 0.0f;
        for (int j = 0; j < nkv; ++j) {
            float wj = __shfl(w, j, 32);
            const float* vptr;
            if (j < WIN_) {
                int p = s + j - (WIN_ / 2);
                vptr = (p >= 0 && p < S) ? Vx + ((size_t)b * S + p) * Din : bv;
            } else if (j == WIN_) {
                vptr = Vb + (size_t)s * Din;
            } else {
                vptr = Va + (size_t)s * Din;
            }
            if (lane < hd) o += wj * vptr[h * hd + lane];
        }
        if (lane < hd) O[(size_t)pos * Din + h * hd + lane] = o;
    }
}

// ---------------------------------------------------------------------------
// DTW prep: per position p, squared norms of fe/fp rows and banded Gram
// G[pos][d+8] = fe[b,p] . fp[b,p+d], d in [-8,8]. One wave per position.
// ---------------------------------------------------------------------------
__global__ void dtw_prep(const float* __restrict__ fe, const float* __restrict__ fp,
                         float* __restrict__ ne, float* __restrict__ np_,
                         float* __restrict__ G, int S, int E) {
    int wave = threadIdx.x >> 5;
    int lane = threadIdx.x & 31;
    int pos = blockIdx.x * (blockDim.x >> 5) + wave;
    if (pos >= BS_) return;
    int b = pos / S, p = pos % S;
    const float* fer = fe + (size_t)pos * E;
    float se = 0.0f, sp = 0.0f;
    for (int e = lane; e < E; e += 32) {
        float a = fer[e];          se += a * a;
        float c = fp[(size_t)pos * E + e]; sp += c * c;
    }
    se = wave_sum(se); sp = wave_sum(sp);
    if (lane == 0) { ne[pos] = se; np_[pos] = sp; }
    for (int d = -8; d <= 8; ++d) {
        int q = p + d;
        float g = 0.0f;
        if (q >= 0 && q < S) {
            const float* fpr = fp + ((size_t)b * S + q) * E;
            for (int e = lane; e < E; e += 32) g += fer[e] * fpr[e];
            g = wave_sum(g);
        }
        if (lane == 0) G[(size_t)pos * 17 + (d + 8)] = g;
    }
}

// ---------------------------------------------------------------------------
// Soft-DTW (gamma=1) over the 9x9 window cost matrix. One thread per (b,s).
// ---------------------------------------------------------------------------
__global__ void sdtw_kernel(const float* __restrict__ ne, const float* __restrict__ np_,
                            const float* __restrict__ G, float* __restrict__ w1, int S) {
    int pos = blockIdx.x * blockDim.x + threadIdx.x;
    if (pos >= BS_) return;
    int b = pos / S, s = pos % S;
    float prev[WIN_ + 1], cur[WIN_ + 1];
    prev[0] = 0.0f;
    #pragma unroll
    for (int j = 1; j <= WIN_; ++j) prev[j] = BIGF;
    for (int i = 1; i <= WIN_; ++i) {
        cur[0] = BIGF;
        int p = s + (i - 1) - (WIN_ / 2);
        bool pv = (p >= 0 && p < S);
        float nep = pv ? ne[(size_t)b * S + p] : 0.0f;
        for (int j = 1; j <= WIN_; ++j) {
            int q = s + (j - 1) - (WIN_ / 2);
            bool qv = (q >= 0 && q < S);
            float npq = qv ? np_[(size_t)b * S + q] : 0.0f;
            float g = (pv && qv) ? G[((size_t)b * S + p) * 17 + (q - p + 8)] : 0.0f;
            float c = sqrtf(fmaxf(nep + npq - 2.0f * g, 1e-12f));
            float x0 = prev[j], x1 = cur[j - 1], x2 = prev[j - 1];
            float mn = fminf(x0, fminf(x1, x2));
            float sm = mn - logf(__expf(mn - x0) + __expf(mn - x1) + __expf(mn - x2));
            cur[j] = c + sm;
        }
        #pragma unroll
        for (int j = 0; j <= WIN_; ++j) prev[j] = cur[j];
    }
    w1[pos] = prev[WIN_];
}

// ---------------------------------------------------------------------------
// Per-batch min/max normalization -> w3 weights. One block per batch.
// ---------------------------------------------------------------------------
__global__ void weights_kernel(const float* __restrict__ w1, float* __restrict__ w3, int S) {
    __shared__ float sh[256];
    __shared__ float s_mn, s_mx, s_sum;
    int b = blockIdx.x, t = threadIdx.x;
    float mn = 1e30f, mx = -1e30f;
    for (int s = t; s < S; s += 256) {
        float v = w1[(size_t)b * S + s];
        mn = fminf(mn, v); mx = fmaxf(mx, v);
    }
    sh[t] = mn; __syncthreads();
    for (int o = 128; o; o >>= 1) { if (t < o) sh[t] = fminf(sh[t], sh[t + o]); __syncthreads(); }
    if (t == 0) s_mn = sh[0];
    __syncthreads();
    sh[t] = mx; __syncthreads();
    for (int o = 128; o; o >>= 1) { if (t < o) sh[t] = fmaxf(sh[t], sh[t + o]); __syncthreads(); }
    if (t == 0) s_mx = sh[0];
    __syncthreads();
    float denom = s_mx - s_mn + 1e-8f;
    float sum = 0.0f;
    for (int s = t; s < S; s += 256)
        sum += 1.0f - (w1[(size_t)b * S + s] - s_mn) / denom;
    sh[t] = sum; __syncthreads();
    for (int o = 128; o; o >>= 1) { if (t < o) sh[t] += sh[t + o]; __syncthreads(); }
    if (t == 0) s_sum = sh[0];
    __syncthreads();
    for (int s = t; s < S; s += 256)
        w3[(size_t)b * S + s] = (1.0f - (w1[(size_t)b * S + s] - s_mn) / denom) / s_sum;
}

// ---------------------------------------------------------------------------
// Weighted pooling: comb[b, half*E + e] = sum_s src[b,s,e] * w3[b,s]
// ---------------------------------------------------------------------------
__global__ void final_sum(const float* __restrict__ fe, const float* __restrict__ fp,
                          const float* __restrict__ w3, float* __restrict__ comb,
                          int S, int E) {
    int b = blockIdx.x, half = blockIdx.y, e = threadIdx.x;
    const float* src = half ? fp : fe;
    float acc = 0.0f;
    for (int s = 0; s < S; ++s)
        acc += src[((size_t)b * S + s) * E + e] * w3[(size_t)b * S + s];
    comb[((size_t)b * 2 + half) * E + e] = acc;
}

// ---------------------------------------------------------------------------
// Classifier: out[b,c] = comb[b,:] @ Wc[:,c] + bc[c]
// ---------------------------------------------------------------------------
__global__ void classifier(const float* __restrict__ comb, const float* __restrict__ Wc,
                           const float* __restrict__ bc, float* __restrict__ out, int Kc) {
    int idx = blockIdx.x * blockDim.x + threadIdx.x;
    if (idx >= BB * NCLS) return;
    int b = idx / NCLS, c = idx % NCLS;
    float acc = bc[c];
    for (int k = 0; k < Kc; ++k) acc += comb[(size_t)b * Kc + k] * Wc[(size_t)k * NCLS + c];
    out[idx] = acc;
}

// ---------------------------------------------------------------------------
// Host orchestration
// ---------------------------------------------------------------------------
extern "C" void kernel_launch(void* const* d_in, const int* in_sizes, int n_in,
                              void* d_out, int out_size, void* d_ws, size_t ws_size,
                              hipStream_t stream) {
    (void)in_sizes; (void)n_in; (void)out_size; (void)ws_size;
    // Input map (setup_inputs dict order, attn dicts in Wq,bq,Wk,bk,Wv,bv,Wo,bo order)
    const float* eeg = (const float*)d_in[0];
    const float* pps = (const float*)d_in[1];
    const float* ie[8]; for (int i = 0; i < 8; ++i) ie[i] = (const float*)d_in[2 + i];
    const float* ip[8]; for (int i = 0; i < 8; ++i) ip[i] = (const float*)d_in[10 + i];
    const float* ep[8]; for (int i = 0; i < 8; ++i) ep[i] = (const float*)d_in[18 + i];
    const float* pe_[8]; for (int i = 0; i < 8; ++i) pe_[i] = (const float*)d_in[26 + i];
    const float* a_e = (const float*)d_in[34];
    const float* b_e = (const float*)d_in[35];
    const float* a_p = (const float*)d_in[36];
    const float* b_p = (const float*)d_in[37];
    const float* Wc  = (const float*)d_in[38];
    const float* bc  = (const float*)d_in[39];
    float* out = (float*)d_out;

    // --- workspace bump allocator (256B aligned) ---
    char* ws = (char*)d_ws;
    size_t off = 0;
    auto alloc = [&](size_t bytes) -> void* {
        void* p = ws + off;
        off = (off + bytes + 255) & ~(size_t)255;
        return p;
    };
    auto fbuf = [&](size_t elems) { return (float*)alloc(elems * sizeof(float)); };
    auto ubuf = [&](size_t elems) { return (uint32_t*)alloc(elems * sizeof(uint32_t)); };

    // packed weights (uint32 count = K*N/2)
    uint32_t* wp_ie_q = ubuf(256 * 256 / 2);
    uint32_t* wp_ie_k = ubuf(256 * 256 / 2);
    uint32_t* wp_ie_v = ubuf(256 * 256 / 2);
    uint32_t* wp_ie_o = ubuf(256 * 256 / 2);
    uint32_t* wp_ip_q = ubuf(128 * 128 / 2);
    uint32_t* wp_ip_k = ubuf(128 * 128 / 2);
    uint32_t* wp_ip_v = ubuf(128 * 128 / 2);
    uint32_t* wp_ip_o = ubuf(128 * 256 / 2);
    uint32_t* wp_ep_q = ubuf(256 * 256 / 2);
    uint32_t* wp_ep_k = ubuf(256 * 256 / 2);
    uint32_t* wp_ep_v = ubuf(256 * 256 / 2);
    uint32_t* wp_ep_o = ubuf(256 * 256 / 2);
    uint32_t* wp_pe_q = ubuf(256 * 256 / 2);
    uint32_t* wp_pe_k = ubuf(256 * 256 / 2);
    uint32_t* wp_pe_v = ubuf(256 * 256 / 2);
    uint32_t* wp_pe_o = ubuf(256 * 256 / 2);

    // token projections
    float* qa_e = fbuf(SS * 256); float* ka_e = fbuf(SS * 256); float* va_e = fbuf(SS * 256);
    float* kb_e = fbuf(SS * 256); float* vb_e = fbuf(SS * 256);
    float* qa_p = fbuf(SS * 128); float* ka_p = fbuf(SS * 128); float* va_p = fbuf(SS * 128);
    float* kb_p = fbuf(SS * 128); float* vb_p = fbuf(SS * 128);

    // dtw / pooling scratch
    float* ne  = fbuf(BS_);
    float* np_ = fbuf(BS_);
    float* Gb  = fbuf((size_t)BS_ * 17);
    float* w1  = fbuf(BS_);
    float* w3  = fbuf(BS_);
    float* comb = fbuf(BB * 2 * EMB);

    // big reusable slots (BS x 256 f32 = 16MB each)
    float* slot[8];
    for (int i = 0; i < 8; ++i) slot[i] = fbuf((size_t)BS_ * 256);

    // --- helpers ---
    auto repack = [&](const float* W, uint32_t* Wp, int K, int N, int hd) {
        int total = (N / 16) * (K / 32) * 256;
        repack_weight<<<(total + 255) / 256, 256, 0, stream>>>(W, Wp, K, N, hd);
    };
    auto gemm = [&](const float* X, const uint32_t* Wp, const float* bias, float* Y,
                    int M, int K, int N) {
        if (K == 256) {          // N == 256: 4 waves x 4 tiles = 256 cols per block
            dim3 g(M / 16, N / 256);
            wmma_gemm<256, 4><<<g, 128, 0, stream>>>(X, Wp, bias, Y, N);
        } else if (N == 256) {   // K=128 -> N=256 output projection
            dim3 g(M / 16, 1);
            wmma_gemm<128, 4><<<g, 128, 0, stream>>>(X, Wp, bias, Y, N);
        } else {                 // K=128, N=128: 4 waves x 2 tiles = 128 cols
            dim3 g(M / 16, 1);
            wmma_gemm<128, 2><<<g, 128, 0, stream>>>(X, Wp, bias, Y, N);
        }
    };
    auto attn = [&](const float* Q, const float* Kx, const float* Vx,
                    const float* Ka, const float* Va, const float* Kb, const float* Vb,
                    const float* bk, const float* bv, float* O, int Din, int hd, int intra) {
        attn_kernel<<<BS_ / 8, 256, 0, stream>>>(Q, Kx, Vx, Ka, Va, Kb, Vb, bk, bv,
                                                 O, Din, hd, intra, SS);
    };

    // --- 1. repack all projection weights ---
    repack(ie[0], wp_ie_q, 256, 256, 32); repack(ie[2], wp_ie_k, 256, 256, 32);
    repack(ie[4], wp_ie_v, 256, 256, 32); repack(ie[6], wp_ie_o, 256, 256, 0);
    repack(ip[0], wp_ip_q, 128, 128, 16); repack(ip[2], wp_ip_k, 128, 128, 16);
    repack(ip[4], wp_ip_v, 128, 128, 16); repack(ip[6], wp_ip_o, 128, 256, 0);
    repack(ep[0], wp_ep_q, 256, 256, 32); repack(ep[2], wp_ep_k, 256, 256, 32);
    repack(ep[4], wp_ep_v, 256, 256, 32); repack(ep[6], wp_ep_o, 256, 256, 0);
    repack(pe_[0], wp_pe_q, 256, 256, 32); repack(pe_[2], wp_pe_k, 256, 256, 32);
    repack(pe_[4], wp_pe_v, 256, 256, 32); repack(pe_[6], wp_pe_o, 256, 256, 0);

    // --- 2. token projections (batch independent, S rows) ---
    gemm(a_e, wp_ie_q, ie[1], qa_e, SS, 256, 256);
    gemm(a_e, wp_ie_k, ie[3], ka_e, SS, 256, 256);
    gemm(a_e, wp_ie_v, ie[5], va_e, SS, 256, 256);
    gemm(b_e, wp_ie_k, ie[3], kb_e, SS, 256, 256);
    gemm(b_e, wp_ie_v, ie[5], vb_e, SS, 256, 256);
    gemm(a_p, wp_ip_q, ip[1], qa_p, SS, 128, 128);
    gemm(a_p, wp_ip_k, ip[3], ka_p, SS, 128, 128);
    gemm(a_p, wp_ip_v, ip[5], va_p, SS, 128, 128);
    gemm(b_p, wp_ip_k, ip[3], kb_p, SS, 128, 128);
    gemm(b_p, wp_ip_v, ip[5], vb_p, SS, 128, 128);

    // --- 3. intra EEG ---
    gemm(eeg, wp_ie_k, ie[3], slot[0], BS_, 256, 256);   // Kx
    gemm(eeg, wp_ie_v, ie[5], slot[1], BS_, 256, 256);   // Vx
    attn(qa_e, slot[0], slot[1], ka_e, va_e, kb_e, vb_e, ie[3], ie[5],
         slot[2], 256, 32, 1);                            // o
    gemm(slot[2], wp_ie_o, ie[7], slot[3], BS_, 256, 256); // ee -> slot3

    // --- 4. intra PPS ---
    gemm(pps, wp_ip_k, ip[3], slot[0], BS_, 128, 128);
    gemm(pps, wp_ip_v, ip[5], slot[1], BS_, 128, 128);
    attn(qa_p, slot[0], slot[1], ka_p, va_p, kb_p, vb_p, ip[3], ip[5],
         slot[2], 128, 16, 1);
    gemm(slot[2], wp_ip_o, ip[7], slot[4], BS_, 128, 256); // pe -> slot4

    // --- 5. cross e2p (q=ee, kv=window of pe) -> fe ---
    gemm(slot[3], wp_ep_q, ep[1], slot[0], BS_, 256, 256);  // Qe
    gemm(slot[4], wp_ep_k, ep[3], slot[1], BS_, 256, 256);  // Kp
    gemm(slot[4], wp_ep_v, ep[5], slot[2], BS_, 256, 256);  // Vp
    attn(slot[0], slot[1], slot[2], nullptr, nullptr, nullptr, nullptr,
         ep[3], ep[5], slot[7], 256, 32, 0);
    gemm(slot[7], wp_ep_o, ep[7], slot[5], BS_, 256, 256);  // fe -> slot5

    // --- 6. cross p2e (q=pe, kv=window of ee) -> fp ---
    gemm(slot[4], wp_pe_q, pe_[1], slot[0], BS_, 256, 256); // Qp
    gemm(slot[3], wp_pe_k, pe_[3], slot[1], BS_, 256, 256); // Ke
    gemm(slot[3], wp_pe_v, pe_[5], slot[2], BS_, 256, 256); // Ve
    attn(slot[0], slot[1], slot[2], nullptr, nullptr, nullptr, nullptr,
         pe_[3], pe_[5], slot[7], 256, 32, 0);
    gemm(slot[7], wp_pe_o, pe_[7], slot[6], BS_, 256, 256); // fp -> slot6

    // --- 7. soft-DTW weighting ---
    dtw_prep<<<BS_ / 8, 256, 0, stream>>>(slot[5], slot[6], ne, np_, Gb, SS, EMB);
    sdtw_kernel<<<BS_ / 256, 256, 0, stream>>>(ne, np_, Gb, w1, SS);
    weights_kernel<<<BB, 256, 0, stream>>>(w1, w3, SS);

    // --- 8. weighted pooling + classifier ---
    final_sum<<<dim3(BB, 2), EMB, 0, stream>>>(slot[5], slot[6], w3, comb, SS, EMB);
    classifier<<<1, 128, 0, stream>>>(comb, Wc, bc, out, 2 * EMB);
}
